// PVConvBlock_20194936226220
// MI455X (gfx1250) — matrix operand order repaired and checked by reading, not compiled, not run on hardware
//
#include <hip/hip_runtime.h>

typedef __attribute__((ext_vector_type(16))) _Float16 v16h;
typedef __attribute__((ext_vector_type(8)))  _Float16 v8h;
typedef __attribute__((ext_vector_type(8)))  float    v8f;

#define DIMV 64
#define CCH  64
#define NPTS 200000
#define EPS_BN 1e-3f
#define VOX   (DIMV*DIMV*DIMV)          // 262144 voxels
#define VOXC  (VOX*CCH)                 // 16777216 elements

// ---------------------------------------------------------------------------
// workspace layout (bytes, all offsets 256B aligned)
//   [0)              mlp_out  f32  N*64        = 51,200,000
//   [ACCUM_OFF)      accum    f32  VOXC        = 67,108,864   (reused as grid2 f16)
//   [GRID0_OFF)      grid0    f16  VOXC        = 33,554,432
//   [GRID1_OFF)      grid1    f16  VOXC        = 33,554,432
// ---------------------------------------------------------------------------
#define MLP_OFF    0
#define ACCUM_OFF  51200000UL
#define GRID0_OFF  118308864UL
#define GRID1_OFF  151863296UL

// ======================= zero the f32 accumulation grid =====================
__global__ void zero_accum_kernel(float* __restrict__ acc) {
    int idx = blockIdx.x * blockDim.x + threadIdx.x;
    if (idx < VOXC) acc[idx] = 0.0f;
}

// ======================= scatter-add voxelization ===========================
__global__ void scatter_kernel(const float* __restrict__ inputs,
                               const int* __restrict__ vidx,
                               float* __restrict__ acc) {
    int idx = blockIdx.x * blockDim.x + threadIdx.x;
    if (idx >= NPTS * CCH) return;
    int i = idx >> 6;
    int c = idx & 63;
    int ix = vidx[i * 3 + 0];
    int iy = vidx[i * 3 + 1];
    int iz = vidx[i * 3 + 2];
    atomicAdd(&acc[(((ix * DIMV) + iy) * DIMV + iz) * CCH + c], inputs[idx]);
}

// ============== per-voxel mean normalization + f32 -> f16 ===================
__global__ void normalize_kernel(const float* __restrict__ acc,
                                 const float* __restrict__ ppv_inv,
                                 _Float16* __restrict__ grid0) {
    int idx = blockIdx.x * blockDim.x + threadIdx.x;
    if (idx >= VOXC) return;
    int v = idx >> 6;
    grid0[idx] = (_Float16)(acc[idx] * ppv_inv[v]);
}

// ===========================================================================
// Conv3D 3x3x3 C=64->64 as implicit GEMM on WMMA.
// One wave = TWO z-lines of 16 output voxels x 64 output channels
//          = 8 accumulator tiles of 16x16 f32 (64 VGPRs).
// Each LDS B-fragment load feeds two WMMAs (2:1 wmma:ds ratio).
// Full weight tensor (27*64*64 f16 = 216KB) staged in LDS, pre-swizzled into
// B-fragment layout so each B operand is one contiguous 32B LDS read.
// Taps fully unrolled (kx/ky/kz compile-time) -> no div/mod SALU, deep
// software pipelining of global A loads / LDS B loads under the WMMAs.
// Fused epilogue: +bias, ReLU, inference BN, write f16.
// ===========================================================================
__global__ void __launch_bounds__(256)
conv3d_wmma_kernel(const _Float16* __restrict__ gin,
                   _Float16* __restrict__ gout,
                   const float* __restrict__ wk,   // [27][64][64]
                   const float* __restrict__ cb,
                   const float* __restrict__ g,
                   const float* __restrict__ be,
                   const float* __restrict__ m,
                   const float* __restrict__ vv) {
    extern __shared__ __align__(128) _Float16 smem[];  // 27*2*4*32*16 halves = 216KB

    const int tid = threadIdx.x;                // 256 threads = 8 waves
    // ---- stage + swizzle weights into B-fragment layout -------------------
    // slot s = (((t*2 + kb)*4 + nt)*32 + lane); each slot = 16 contiguous halves
    for (int s = tid; s < 27 * 2 * 4 * 32; s += 256) {
        int l    = s & 31;
        int nt   = (s >> 5) & 3;
        int kb   = (s >> 7) & 1;
        int t    = s >> 8;
        int cout = nt * 16 + (l & 15);
        int cin0 = kb * 32 + ((l >> 4) * 16);
        v16h tmp;
#pragma unroll
        for (int e = 0; e < 16; ++e)
            tmp[e] = (_Float16)wk[((t * CCH) + (cin0 + e)) * CCH + cout];
        *(v16h*)(smem + (size_t)s * 16) = tmp;
    }
    __syncthreads();

    const int lane = tid & 31;
    const int wave = tid >> 5;
    const int wid  = blockIdx.x * 8 + wave;     // 8192 waves: 64 x 64 x 2
    const int x  = wid >> 7;
    const int y  = (wid >> 1) & 63;
    const int z0 = (wid & 1) * 32;              // wave covers z0..z0+31

    const int mlane = lane & 15;                // A-row / C-col select
    const int hsel  = lane >> 4;                // K-half select

    v8f acc[2][4];
#pragma unroll
    for (int mt = 0; mt < 2; ++mt)
#pragma unroll
        for (int nt = 0; nt < 4; ++nt) acc[mt][nt] = {};

#pragma unroll
    for (int kx = 0; kx < 3; ++kx) {
        const int xin = x + kx - 1;
        if (xin < 0 || xin >= DIMV) continue;   // uniform -> scalar branch
#pragma unroll
        for (int ky = 0; ky < 3; ++ky) {
            const int yin = y + ky - 1;
            if (yin < 0 || yin >= DIMV) continue;
            const size_t pbase = ((size_t)xin * DIMV + yin) * DIMV;
#pragma unroll
            for (int kz = 0; kz < 3; ++kz) {
                const int t  = (kx * 3 + ky) * 3 + kz;
                const int dz = kz - 1;
#pragma unroll
                for (int kb = 0; kb < 2; ++kb) {
                    const int c0 = kb * 32 + hsel * 16;
                    // ---- two A fragments (two z-lines), contiguous k = 16*hsel + e order --
                    v16h a[2];
#pragma unroll
                    for (int mt = 0; mt < 2; ++mt) {
                        const int zin = z0 + mt * 16 + mlane + dz;
                        v8h lo = {};
                        v8h hi = {};
                        if (zin >= 0 && zin < DIMV) {
                            const _Float16* p = gin + (pbase + (size_t)zin) * CCH + c0;
                            lo = *(const v8h*)p;
                            hi = *(const v8h*)(p + 8);
                        }
#pragma unroll
                        for (int e = 0; e < 8; ++e) {
                            a[mt][e]     = lo[e];
                            a[mt][e + 8] = hi[e];
                        }
                    }
                    // ---- 4 N-tiles of B; each feeds two WMMAs ----
#pragma unroll
                    for (int nt = 0; nt < 4; ++nt) {
                        v16h b = *(const v16h*)(
                            smem + (size_t)((((t * 2 + kb) * 4 + nt) * 32 + lane)) * 16);
                        acc[0][nt] = __builtin_amdgcn_wmma_f32_16x16x32_f16(
                            false, a[0], false, b, (short)0, acc[0][nt], false, false);
                        acc[1][nt] = __builtin_amdgcn_wmma_f32_16x16x32_f16(
                            false, a[1], false, b, (short)0, acc[1][nt], false, false);
                    }
                }
            }
        }
    }

    // ---- fused bias + ReLU + BN epilogue, store f16 -----------------------
#pragma unroll
    for (int nt = 0; nt < 4; ++nt) {
        const int c = nt * 16 + mlane;
        const float scale = g[c] * rsqrtf(vv[c] + EPS_BN);
        const float shift = be[c] - m[c] * scale;
        const float bias  = cb[c];
#pragma unroll
        for (int mt = 0; mt < 2; ++mt) {
#pragma unroll
            for (int r = 0; r < 8; ++r) {
                const int zr = z0 + mt * 16 + hsel * 8 + r;
                float val = acc[mt][nt][r] + bias;
                val = fmaxf(val, 0.0f);
                val = val * scale + shift;
                gout[((((size_t)x * DIMV + y) * DIMV + zr) * CCH) + c] = (_Float16)val;
            }
        }
    }
}

// ===========================================================================
// Point MLP: Dense(64->64) + ReLU + BN as WMMA GEMM.
// N = 200000 = 12500 exact 16-row tiles; one wave per tile, 8 waves/block.
// ===========================================================================
__global__ void mlp_wmma_kernel(const float* __restrict__ inp,
                                const float* __restrict__ w,     // [64][64]
                                const float* __restrict__ bmlp,
                                const float* __restrict__ g,
                                const float* __restrict__ be,
                                const float* __restrict__ m,
                                const float* __restrict__ vv,
                                float* __restrict__ mlp_out) {
    __shared__ _Float16 wsm[2 * 4 * 32 * 16] __attribute__((aligned(128)));  // 8KB

    const int tid = threadIdx.x;                // 256 threads, one slot each
    {
        int l    = tid & 31;
        int nt   = (tid >> 5) & 3;
        int kb   = (tid >> 7) & 1;
        int cout = nt * 16 + (l & 15);
        int cin0 = kb * 32 + ((l >> 4) * 16);
        v16h tmp;
#pragma unroll
        for (int e = 0; e < 16; ++e)
            tmp[e] = (_Float16)w[(cin0 + e) * CCH + cout];
        *(v16h*)(wsm + tid * 16) = tmp;
    }
    __syncthreads();

    const int lane = tid & 31;
    const int wave = tid >> 5;
    const int tile = blockIdx.x * 8 + wave;
    if (tile >= NPTS / 16) return;

    const int mlane = lane & 15;
    const int hsel  = lane >> 4;
    const int row   = tile * 16 + mlane;

    v8f acc[4];
#pragma unroll
    for (int nt = 0; nt < 4; ++nt) acc[nt] = {};

#pragma unroll
    for (int kb = 0; kb < 2; ++kb) {
        const int c0 = kb * 32 + hsel * 16;
        v8f flo = *(const v8f*)(inp + (size_t)row * CCH + c0);
        v8f fhi = *(const v8f*)(inp + (size_t)row * CCH + c0 + 8);
        v16h a;
#pragma unroll
        for (int e = 0; e < 8; ++e) {
            a[e]     = (_Float16)flo[e];
            a[e + 8] = (_Float16)fhi[e];
        }
#pragma unroll
        for (int nt = 0; nt < 4; ++nt) {
            v16h b = *(const v16h*)(wsm + ((kb * 4 + nt) * 32 + lane) * 16);
            acc[nt] = __builtin_amdgcn_wmma_f32_16x16x32_f16(
                false, a, false, b, (short)0, acc[nt], false, false);
        }
    }

#pragma unroll
    for (int nt = 0; nt < 4; ++nt) {
        const int c = nt * 16 + mlane;
        const float scale = g[c] * rsqrtf(vv[c] + EPS_BN);
        const float shift = be[c] - m[c] * scale;
        const float bias  = bmlp[c];
#pragma unroll
        for (int r = 0; r < 8; ++r) {
            const int rr = tile * 16 + hsel * 8 + r;
            float val = acc[nt][r] + bias;
            val = fmaxf(val, 0.0f);
            mlp_out[(size_t)rr * CCH + c] = val * scale + shift;
        }
    }
}

// ================= trilinear devoxelize + residual add ======================
__global__ void gather_kernel(const _Float16* __restrict__ grid2,
                              const float* __restrict__ pc,
                              const float* __restrict__ mlp_out,
                              float* __restrict__ out) {
    int idx = blockIdx.x * blockDim.x + threadIdx.x;
    if (idx >= NPTS * CCH) return;
    int i = idx >> 6;
    int c = idx & 63;

    float px = fminf(fmaxf(pc[i * 3 + 0], 0.0f), (float)(DIMV - 1));
    float py = fminf(fmaxf(pc[i * 3 + 1], 0.0f), (float)(DIMV - 1));
    float pz = fminf(fmaxf(pc[i * 3 + 2], 0.0f), (float)(DIMV - 1));
    float fx = floorf(px), fy = floorf(py), fz = floorf(pz);
    float tx = px - fx, ty = py - fy, tz = pz - fz;
    int x0 = (int)fx, y0 = (int)fy, z0 = (int)fz;
    int x1 = min(x0 + 1, DIMV - 1);
    int y1 = min(y0 + 1, DIMV - 1);
    int z1 = min(z0 + 1, DIMV - 1);

    float val = 0.0f;
#pragma unroll
    for (int dx = 0; dx < 2; ++dx) {
        int ix = dx ? x1 : x0;
        float wx = dx ? tx : 1.0f - tx;
#pragma unroll
        for (int dy = 0; dy < 2; ++dy) {
            int iy = dy ? y1 : y0;
            float wy = dy ? ty : 1.0f - ty;
#pragma unroll
            for (int dz = 0; dz < 2; ++dz) {
                int iz = dz ? z1 : z0;
                float wz = dz ? tz : 1.0f - tz;
                float v = (float)grid2[((((size_t)ix * DIMV + iy) * DIMV + iz) * CCH) + c];
                val += wx * wy * wz * v;
            }
        }
    }
    out[idx] = mlp_out[idx] + val;
}

// ===========================================================================
extern "C" void kernel_launch(void* const* d_in, const int* in_sizes, int n_in,
                              void* d_out, int out_size, void* d_ws, size_t ws_size,
                              hipStream_t stream) {
    const float* inputs  = (const float*)d_in[0];
    const float* pcoords = (const float*)d_in[1];
    const int*   vidx    = (const int*)d_in[2];
    const float* ppv_inv = (const float*)d_in[3];
    const float* w_mlp   = (const float*)d_in[4];
    const float* b_mlp   = (const float*)d_in[5];
    const float* g0  = (const float*)d_in[6];
    const float* be0 = (const float*)d_in[7];
    const float* m0  = (const float*)d_in[8];
    const float* v0  = (const float*)d_in[9];
    const float* k1  = (const float*)d_in[10];
    const float* cb1 = (const float*)d_in[11];
    const float* g1  = (const float*)d_in[12];
    const float* be1 = (const float*)d_in[13];
    const float* m1  = (const float*)d_in[14];
    const float* v1  = (const float*)d_in[15];
    const float* k2  = (const float*)d_in[16];
    const float* cb2 = (const float*)d_in[17];
    const float* g2  = (const float*)d_in[18];
    const float* be2 = (const float*)d_in[19];
    const float* m2  = (const float*)d_in[20];
    const float* v2  = (const float*)d_in[21];

    char* ws = (char*)d_ws;
    float*     mlp_out = (float*)(ws + MLP_OFF);
    float*     accum   = (float*)(ws + ACCUM_OFF);
    _Float16*  grid0   = (_Float16*)(ws + GRID0_OFF);
    _Float16*  grid1   = (_Float16*)(ws + GRID1_OFF);
    _Float16*  grid2   = (_Float16*)(ws + ACCUM_OFF);  // reuse accum after conv1
    float* out = (float*)d_out;

    const int convLdsBytes = 27 * 2 * 4 * 32 * 16 * 2;  // 221184 B (fits 320KB WGP LDS)

    // point-branch MLP GEMM (independent of voxel branch)
    mlp_wmma_kernel<<<(NPTS / 16 + 7) / 8, 256, 0, stream>>>(
        inputs, w_mlp, b_mlp, g0, be0, m0, v0, mlp_out);

    // voxel branch
    zero_accum_kernel<<<VOXC / 256, 256, 0, stream>>>(accum);
    scatter_kernel<<<(NPTS * CCH) / 256, 256, 0, stream>>>(inputs, vidx, accum);
    normalize_kernel<<<VOXC / 256, 256, 0, stream>>>(accum, ppv_inv, grid0);

    // voxel branch convs: 1024 blocks x 8 waves = 8192 waves (64*64*2 z-pairs)
    conv3d_wmma_kernel<<<1024, 256, convLdsBytes, stream>>>(
        grid0, grid1, k1, cb1, g1, be1, m1, v1);
    conv3d_wmma_kernel<<<1024, 256, convLdsBytes, stream>>>(
        grid1, grid2, k2, cb2, g2, be2, m2, v2);

    // trilinear devoxelize + residual
    gather_kernel<<<(NPTS * CCH) / 256, 256, 0, stream>>>(
        grid2, pcoords, mlp_out, out);
}
